// Simple_Generator_17179869784
// MI455X (gfx1250) — matrix operand order repaired and checked by reading, not compiled, not run on hardware
//
#include <hip/hip_runtime.h>
#include <hip/hip_bf16.h>

typedef _Float16 v16h __attribute__((ext_vector_type(16)));
typedef _Float16 v8h  __attribute__((ext_vector_type(8)));
typedef float    v8f  __attribute__((ext_vector_type(8)));
typedef int      v4i  __attribute__((ext_vector_type(4)));

// ---------------- CDNA5 async global->LDS path (guarded) ------------------
#ifndef USE_ASYNC_LDS
#if defined(__has_builtin)
#if __has_builtin(__builtin_amdgcn_global_load_async_to_lds_b128)
#define USE_ASYNC_LDS 1
#endif
#endif
#endif
#ifndef USE_ASYNC_LDS
#define USE_ASYNC_LDS 0
#endif

typedef __attribute__((address_space(1))) v4i* gv4i_ptr;  // global
typedef __attribute__((address_space(3))) v4i* lv4i_ptr;  // LDS

__device__ __forceinline__ void copy16_g2s(void* dst, const void* src) {
#if USE_ASYNC_LDS
  __builtin_amdgcn_global_load_async_to_lds_b128(
      (gv4i_ptr)src, (lv4i_ptr)dst, 0, 0);
#else
  *(v4i*)dst = *(const v4i*)src;
#endif
}

__device__ __forceinline__ void async_copy_wait() {
#if USE_ASYNC_LDS
#if defined(__has_builtin) && __has_builtin(__builtin_amdgcn_s_wait_asynccnt)
  __builtin_amdgcn_s_wait_asynccnt(0);
#else
  asm volatile("s_wait_asynccnt 0" ::: "memory");
#endif
#endif
}

// ---------------- activation helpers ------------------
__device__ __forceinline__ float celu1(float v) {          // CELU, alpha=1
  return v > 0.f ? v : (__expf(v) - 1.f);
}
__device__ __forceinline__ float selu_act(float v) {
  const float a = 1.6732632423543772f, s = 1.0507009873554805f;
  return v > 0.f ? s * v : s * a * (__expf(v) - 1.f);
}

// =====================================================================
// Kernel 0: per-(b,c) mean / rstd (unbiased std, ddof=1) of x[16,3,256,256]
// =====================================================================
__global__ __launch_bounds__(256) void stats_kernel(const float* __restrict__ x,
                                                    float* __restrict__ stat) {
  __shared__ float ss[256], ss2[256];
  const int bc = blockIdx.x;                       // 0..47
  const float* p = x + (size_t)bc * 65536;
  float s = 0.f, s2 = 0.f;
  for (int i = threadIdx.x; i < 65536; i += 256) {
    float v = p[i];
    s += v; s2 += v * v;
  }
  ss[threadIdx.x] = s; ss2[threadIdx.x] = s2;
  __syncthreads();
  for (int off = 128; off > 0; off >>= 1) {
    if ((int)threadIdx.x < off) {
      ss[threadIdx.x]  += ss[threadIdx.x + off];
      ss2[threadIdx.x] += ss2[threadIdx.x + off];
    }
    __syncthreads();
  }
  if (threadIdx.x == 0) {
    float mean = ss[0] * (1.f / 65536.f);
    float var  = (ss2[0] - 65536.f * mean * mean) * (1.f / 65535.f);
    stat[bc * 2 + 0] = mean;
    stat[bc * 2 + 1] = rsqrtf(var);
  }
}

// =====================================================================
// Kernel 1: fold conv-bias + BN into per-channel scale/shift.
// layers 0..9 = blocks, layer 10 = conv1/bn1.
// =====================================================================
__global__ void fold_bn(const float* __restrict__ conv1_b,
                        const float* __restrict__ bn1_g, const float* __restrict__ bn1_b,
                        const float* __restrict__ bn1_m, const float* __restrict__ bn1_v,
                        const float* __restrict__ blk_b, const float* __restrict__ blk_g,
                        const float* __restrict__ blk_be, const float* __restrict__ blk_m,
                        const float* __restrict__ blk_v,
                        float* __restrict__ scales, float* __restrict__ shifts) {
  const int layer = blockIdx.x;     // 0..10
  const int c = threadIdx.x;        // 0..63
  float g, be, m, v, bias;
  if (layer == 10) {
    g = bn1_g[c]; be = bn1_b[c]; m = bn1_m[c]; v = bn1_v[c]; bias = conv1_b[c];
  } else {
    int o = layer * 64 + c;
    g = blk_g[o]; be = blk_be[o]; m = blk_m[o]; v = blk_v[o]; bias = blk_b[o];
  }
  float s = g * rsqrtf(v + 1e-5f);
  scales[layer * 64 + c] = s;
  shifts[layer * 64 + c] = (bias - m) * s + be;
}

// =====================================================================
// Kernel 2: pack block weights [5][2][64][64][3][3] f32 (OIHW) into
// per-lane B-fragment layout for v_wmma_f32_16x16x32_f16:
//   wpack[layer][tap][kchunk(2)][ntile(4)][lane(32)][i(16)]  (f16)
//   lane 0-15 : N = lane,     K = kchunk*32 + (i&7) + (i>>3)*16      (i = 0..15)
//   lane16-31 : N = lane-16,  K = kchunk*32 + 8 + (i&7) + (i>>3)*16
// =====================================================================
__global__ __launch_bounds__(256) void pack_weights(const float* __restrict__ blk_w,
                                                    _Float16* __restrict__ wpack) {
  int idx = blockIdx.x * 256 + threadIdx.x;
  if (idx >= 10 * 36864) return;
  const int layer = idx / 36864;
  int r = idx % 36864;
  const int i    = r & 15;  r >>= 4;
  const int lane = r & 31;  r >>= 5;
  const int nt   = r & 3;   r >>= 2;
  const int kc   = r & 1;   r >>= 1;
  const int tap  = r;                         // 0..8 = ky*3+kx
  const int n    = nt * 16 + (lane & 15);     // out channel
  const int k    = kc * 32 + (lane >> 4) * 8 + (i & 7) + (i >> 3) * 16;  // in channel
  float wv = blk_w[(((size_t)layer * 64 + n) * 64 + k) * 9 + tap];
  wpack[idx] = (_Float16)wv;
}

// =====================================================================
// Kernel 3: normalize+concat + conv1 (6->64) + BN + SELU -> NHWC f16
// =====================================================================
__global__ __launch_bounds__(256) void conv1_kernel(
    const float* __restrict__ x, const float* __restrict__ w,
    const float* __restrict__ scale, const float* __restrict__ shift,
    const float* __restrict__ stat, _Float16* __restrict__ actOut) {
  __shared__ float sw[64 * 6 * 9];
  for (int i = threadIdx.x; i < 64 * 6 * 9; i += 256) sw[i] = w[i];
  __syncthreads();

  const int id  = blockIdx.x * 256 + threadIdx.x;   // 0..1048575
  const int b   = id >> 16;
  const int pix = id & 65535;
  const int yy  = pix >> 8;
  const int xx  = pix & 255;

  float in[6][9];
  for (int ci = 0; ci < 3; ++ci) {
    const float mean = stat[(b * 3 + ci) * 2 + 0];
    const float rstd = stat[(b * 3 + ci) * 2 + 1];
    const float* xp = x + (size_t)(b * 3 + ci) * 65536;
#pragma unroll
    for (int ky = 0; ky < 3; ++ky)
#pragma unroll
      for (int kx = 0; kx < 3; ++kx) {
        int gy = yy + ky - 1, gx = xx + kx - 1;
        bool ok = ((unsigned)gy < 256u) && ((unsigned)gx < 256u);
        float v = ok ? xp[gy * 256 + gx] : 0.f;
        in[ci][ky * 3 + kx]     = v;
        in[ci + 3][ky * 3 + kx] = ok ? (v - mean) * rstd : 0.f;
      }
  }
  _Float16* op = actOut + (size_t)id * 64;
  for (int oc = 0; oc < 64; ++oc) {
    float acc = 0.f;
    const float* wo = &sw[oc * 54];
#pragma unroll
    for (int c6 = 0; c6 < 6; ++c6)
#pragma unroll
      for (int t = 0; t < 9; ++t)
        acc = fmaf(in[c6][t], wo[c6 * 9 + t], acc);
    acc = acc * scale[oc] + shift[oc];
    op[oc] = (_Float16)selu_act(acc);
  }
}

// =====================================================================
// Kernel 4 (x10): 3x3 conv 64->64 as implicit GEMM with WMMA f16.
// WG = 256 threads (8 wave32). Output tile: 128 spatial x 64 ch.
// Wave w: spatial rows [16w,16w+16), all 4 N-tiles of 16 channels.
// K loop: 9 taps x 2 k-chunks of 32 -> 72 v_wmma per wave.
// Epilogue: scale/shift (folded BN+bias) + CELU -> NHWC f16.
// =====================================================================
__global__ __launch_bounds__(256) void conv64_wmma(
    const _Float16* __restrict__ actIn, _Float16* __restrict__ actOut,
    const _Float16* __restrict__ wpack,
    const float* __restrict__ scale, const float* __restrict__ shift) {
  __shared__ alignas(16) _Float16 sIn[3 * 130 * 64];        // 49,920 B halo tile
  __shared__ alignas(32) _Float16 sW[9 * 2 * 4 * 32 * 16];  // 73,728 B weights

  const int tid = threadIdx.x;
  const int blk = blockIdx.x;
  const int x0 = (blk & 1) * 128;
  const int y  = (blk >> 1) & 255;
  const int b  = blk >> 9;

  // ---- stage packed weights: 4608 x 16B chunks
  for (int i = tid; i < 4608; i += 256)
    copy16_g2s((char*)sW + i * 16, (const char*)wpack + i * 16);

  // ---- stage input halo rows (3 rows x 130 cols x 64ch): 3120 x 16B chunks
  for (int i = tid; i < 3120; i += 256) {
    const int cpart = i & 7;          // 8-channel chunk within pixel
    const int pix   = i >> 3;         // 0..389
    const int xx    = pix % 130;
    const int r     = pix / 130;
    const int gy = y + r - 1;
    const int gx = x0 + xx - 1;
    _Float16* dst = &sIn[(r * 130 + xx) * 64 + cpart * 8];
    if ((unsigned)gy < 256u && (unsigned)gx < 256u) {
      const _Float16* src =
          &actIn[(((size_t)b * 256 + gy) * 256 + gx) * 64 + cpart * 8];
      copy16_g2s(dst, src);
    } else {
      v4i z = {0, 0, 0, 0};
      *(v4i*)dst = z;
    }
  }
  async_copy_wait();
  __syncthreads();

  const int wave = tid >> 5;
  const int lane = tid & 31;
  const int m    = lane & 15;   // A-matrix row (spatial) / D-matrix column index
  const int half = lane >> 4;
  const int lx   = wave * 16 + m;

  v8f acc[4] = {};

  for (int ky = 0; ky < 3; ++ky) {
#pragma unroll
    for (int kx = 0; kx < 3; ++kx) {
      const int tap = ky * 3 + kx;
#pragma unroll
      for (int c = 0; c < 2; ++c) {
        // A fragment per ISA 16-bit A 16x32 layout:
        // lanes 0-15: halves0-7 = K(kbase..+7), halves8-15 = K(kbase+16..+23)
        // lanes16-31: kbase += 8
        union { v16h v; v8h h[2]; } a;
        const _Float16* ap =
            &sIn[(ky * 130 + lx + kx) * 64 + c * 32 + half * 8];
        a.h[0] = *(const v8h*)(ap);
        a.h[1] = *(const v8h*)(ap + 16);
        const _Float16* wp = &sW[((tap * 2 + c) * 4) * 32 * 16 + lane * 16];
#pragma unroll
        for (int nt = 0; nt < 4; ++nt) {
          v16h bfr = *(const v16h*)(wp + nt * 32 * 16);
          acc[nt] = __builtin_amdgcn_wmma_f32_16x16x32_f16(
              false, a.v, false, bfr, (short)0, acc[nt], false, false);
        }
      }
    }
  }

  // ---- epilogue: D layout -> lane gives N = lane&15, row = r + 8*(lane>>4)
  float sc[4], sh[4];
#pragma unroll
  for (int nt = 0; nt < 4; ++nt) {
    int n = nt * 16 + m;
    sc[nt] = scale[n];
    sh[nt] = shift[n];
  }
  const size_t outBase = (((size_t)b * 256 + y) * 256 + x0 + wave * 16) * 64;
#pragma unroll
  for (int nt = 0; nt < 4; ++nt) {
    const int n = nt * 16 + m;
#pragma unroll
    for (int r = 0; r < 8; ++r) {
      float v = acc[nt][r] * sc[nt] + sh[nt];
      v = celu1(v);
      const int p = r + 8 * half;  // spatial row within wave tile
      actOut[outBase + (size_t)p * 64 + n] = (_Float16)v;
    }
  }
}

// =====================================================================
// Kernel 5: conv2 (64->3) + SELU + residual -> fp32 NCHW output
// =====================================================================
__global__ __launch_bounds__(256) void conv2_kernel(
    const _Float16* __restrict__ actIn, const float* __restrict__ w,
    const float* __restrict__ cb, const float* __restrict__ x,
    float* __restrict__ out) {
  __shared__ float sw[9 * 64 * 3];  // [tap][ic][oc]
  for (int i = threadIdx.x; i < 1728; i += 256) {
    int oc = i % 3;
    int tic = i / 3;
    int ic = tic % 64;
    int t  = tic / 64;
    sw[i] = w[(oc * 64 + ic) * 9 + t];
  }
  __syncthreads();

  const int id  = blockIdx.x * 256 + threadIdx.x;
  const int b   = id >> 16;
  const int pix = id & 65535;
  const int yy  = pix >> 8;
  const int xx  = pix & 255;

  float a0 = cb[0], a1 = cb[1], a2 = cb[2];
  for (int ky = 0; ky < 3; ++ky) {
    int gy = yy + ky - 1;
    if ((unsigned)gy >= 256u) continue;
    for (int kx = 0; kx < 3; ++kx) {
      int gx = xx + kx - 1;
      if ((unsigned)gx >= 256u) continue;
      const int t = ky * 3 + kx;
      const _Float16* p = &actIn[(((size_t)b * 256 + gy) * 256 + gx) * 64];
      const float* wt = &sw[t * 64 * 3];
#pragma unroll
      for (int ic8 = 0; ic8 < 8; ++ic8) {
        v8h chunk = *(const v8h*)(p + ic8 * 8);
#pragma unroll
        for (int j = 0; j < 8; ++j) {
          float v = (float)chunk[j];
          const float* ww = &wt[(ic8 * 8 + j) * 3];
          a0 = fmaf(v, ww[0], a0);
          a1 = fmaf(v, ww[1], a1);
          a2 = fmaf(v, ww[2], a2);
        }
      }
    }
  }
  const size_t o0 = (size_t)b * 3 * 65536 + pix;
  out[o0]             = selu_act(a0) + x[o0];
  out[o0 + 65536]     = selu_act(a1) + x[o0 + 65536];
  out[o0 + 2 * 65536] = selu_act(a2) + x[o0 + 2 * 65536];
}

// =====================================================================
extern "C" void kernel_launch(void* const* d_in, const int* in_sizes, int n_in,
                              void* d_out, int out_size, void* d_ws, size_t ws_size,
                              hipStream_t stream) {
  (void)in_sizes; (void)n_in; (void)out_size; (void)ws_size;
  const float* x       = (const float*)d_in[0];
  const float* conv1_w = (const float*)d_in[1];
  const float* conv1_b = (const float*)d_in[2];
  const float* bn1_g   = (const float*)d_in[3];
  const float* bn1_b   = (const float*)d_in[4];
  const float* bn1_m   = (const float*)d_in[5];
  const float* bn1_v   = (const float*)d_in[6];
  const float* blk_w   = (const float*)d_in[7];
  const float* blk_b   = (const float*)d_in[8];
  const float* blk_g   = (const float*)d_in[9];
  const float* blk_be  = (const float*)d_in[10];
  const float* blk_m   = (const float*)d_in[11];
  const float* blk_v   = (const float*)d_in[12];
  const float* conv2_w = (const float*)d_in[13];
  const float* conv2_b = (const float*)d_in[14];
  float* out = (float*)d_out;

  char* ws = (char*)d_ws;
  float*    stat   = (float*)(ws + 0);                       //   384 B
  float*    scales = (float*)(ws + 1024);                    //  2816 B (11*64)
  float*    shifts = (float*)(ws + 4096);                    //  2816 B
  _Float16* wpack  = (_Float16*)(ws + 8192);                 // 737280 B
  _Float16* actA   = (_Float16*)(ws + (1ull << 20));                       // 134 MB
  _Float16* actB   = (_Float16*)(ws + (1ull << 20) + 134217728ull);        // 134 MB

  stats_kernel<<<48, 256, 0, stream>>>(x, stat);
  fold_bn<<<11, 64, 0, stream>>>(conv1_b, bn1_g, bn1_b, bn1_m, bn1_v,
                                 blk_b, blk_g, blk_be, blk_m, blk_v,
                                 scales, shifts);
  pack_weights<<<(10 * 36864 + 255) / 256, 256, 0, stream>>>(blk_w, wpack);
  conv1_kernel<<<4096, 256, 0, stream>>>(x, conv1_w, scales + 640, shifts + 640,
                                         stat, actA);
  _Float16* cur = actA;
  _Float16* nxt = actB;
  for (int l = 0; l < 10; ++l) {
    conv64_wmma<<<8192, 256, 0, stream>>>(cur, nxt, wpack + (size_t)l * 36864,
                                          scales + l * 64, shifts + l * 64);
    _Float16* t = cur; cur = nxt; nxt = t;
  }
  conv2_kernel<<<4096, 256, 0, stream>>>(cur, conv2_w, conv2_b, x, out);
}
